// CausalSelfAttention_62886911148044
// MI455X (gfx1250) — compile-verified, matrix operand order checked by reading, not run on hardware
//
#include <hip/hip_runtime.h>

// ---------------------------------------------------------------------------
// CausalSelfAttention for MI455X (gfx1250, wave32, WMMA f32_16x16x32_f16,
// async global->LDS staging for K/V tiles in the attention kernel)
//   B=2, S=2048, H=12, D=64, DM=768
// ---------------------------------------------------------------------------

typedef __attribute__((ext_vector_type(16))) _Float16 v16h;
typedef __attribute__((ext_vector_type(8)))  _Float16 v8h;
typedef __attribute__((ext_vector_type(8)))  float    v8f;

#define NB   2
#define NS   2048
#define NH   12
#define ND   64
#define NDM  768
#define BS   (NB*NS)          // 4096 rows

__device__ __forceinline__ v8f wmma_f16(v16h a, v16h b, v8f c) {
    return __builtin_amdgcn_wmma_f32_16x16x32_f16(
        /*neg_a=*/false, a, /*neg_b=*/false, b,
        /*c_mod=*/(short)0, c, /*reuse_a=*/false, /*reuse_b=*/false);
}

// Load a 16x32 f16 fragment (A-layout; also valid for B when given B^T row-major).
// Lane l: row = l&15, K-chunks {h*8..h*8+7, 16+h*8..16+h*8+7}, h = l>>4.
__device__ __forceinline__ v16h load_frag_g(const _Float16* p, int ld, int row0, int k0) {
    int lane = threadIdx.x & 31;
    int m  = lane & 15;
    int hh = (lane >> 4) << 3;
    const _Float16* rp = p + (size_t)(row0 + m) * (size_t)ld + k0;
    v8h lo = *(const v8h*)(rp + hh);
    v8h hi = *(const v8h*)(rp + 16 + hh);
    v16h out;
#pragma unroll
    for (int i = 0; i < 8; ++i) { out[i] = lo[i]; out[i + 8] = hi[i]; }
    return out;
}

// Fragment load from an LDS-resident row-major tile (direct array indexing so
// clang keeps addrspace(3) -> ds_load; adjacent halves merge to ds_load_b128).
#define LOAD_FRAG_LDS(dst, arr, ld, row0, k0)                                  \
    {                                                                          \
        int _hh = lh << 3;                                                     \
        _Pragma("unroll")                                                      \
        for (int _i = 0; _i < 8; ++_i) {                                       \
            dst[_i]     = arr[((row0) + fn) * (ld) + (k0) + _hh + _i];         \
            dst[_i + 8] = arr[((row0) + fn) * (ld) + (k0) + 16 + _hh + _i];    \
        }                                                                      \
    }

// Async global -> LDS copy of one 16-byte chunk (per lane), tracked by ASYNCcnt.
__device__ __forceinline__ void async_copy_b128(const _Float16* g, const _Float16* l) {
    unsigned lds_off = (unsigned)(size_t)l;          // addr[31:0] = LDS byte address
    unsigned long long ga = (unsigned long long)(size_t)g;
    asm volatile("global_load_async_to_lds_b128 %0, %1, off"
                 :: "v"(lds_off), "v"(ga) : "memory");
}

__device__ __forceinline__ void wait_async0() {
    asm volatile("s_wait_asynccnt 0x0" ::: "memory");
}

// ---------------------------------------------------------------------------
// Kernel 0: f32 -> f16 conversion
// ---------------------------------------------------------------------------
__global__ void cvt_f16_kernel(const float* __restrict__ src,
                               _Float16* __restrict__ dst, int n) {
    int i = blockIdx.x * blockDim.x + threadIdx.x;
    if (i < n) dst[i] = (_Float16)src[i];
}

// ---------------------------------------------------------------------------
// Kernel 1: fused QKV projection (y = x @ W^T) + 2D RoPE epilogue.
//   grid = (BS/16, DM/128, 3), block = 128 (4 waves, each a 16x32 tile)
// ---------------------------------------------------------------------------
__global__ void __launch_bounds__(128)
qkv_rope_kernel(const _Float16* __restrict__ xh,
                const _Float16* __restrict__ Wqh,
                const _Float16* __restrict__ Wkh,
                const _Float16* __restrict__ Wvh,
                _Float16* __restrict__ qo,
                _Float16* __restrict__ ko,
                _Float16* __restrict__ vT,
                const int* __restrict__ row_ids,
                const int* __restrict__ col_ids) {
    int lane = threadIdx.x & 31;
    int wave = threadIdx.x >> 5;
    int m0 = blockIdx.x * 16;
    int n0 = (blockIdx.y * 4 + wave) * 32;
    int z  = blockIdx.z;
    const _Float16* W = (z == 0) ? Wqh : (z == 1) ? Wkh : Wvh;

    v8f c0 = {}, c1 = {};
#pragma unroll 4
    for (int kk = 0; kk < NDM; kk += 32) {
        v16h a  = load_frag_g(xh, NDM, m0, kk);
        v16h b0 = load_frag_g(W,  NDM, n0, kk);
        v16h b1 = load_frag_g(W,  NDM, n0 + 16, kk);
        c0 = wmma_f16(a, b0, c0);
        c1 = wmma_f16(a, b1, c1);
    }

    int head = n0 >> 6;
    int half = (n0 >> 5) & 1;
    int fn   = lane & 15;
    int lh   = lane >> 4;

    if (z < 2) {
        const int* ids = half ? col_ids : row_ids;
        _Float16* dst = (z == 0) ? qo : ko;
        float invf = __expf(-0.57564627324851f * (float)fn); // 10000^(-fn/16)
#pragma unroll
        for (int r = 0; r < 8; ++r) {
            int m   = r + 8 * lh;
            int row = m0 + m;
            int b   = row >> 11;
            int s   = row & (NS - 1);
            float ang = (float)ids[s] * invf;
            float cs = __cosf(ang), sn = __sinf(ang);
            float o0 = c0[r] * cs - c1[r] * sn;
            float o1 = c1[r] * cs + c0[r] * sn;
            size_t base = (((size_t)b * NH + head) * NS + s) * ND + (n0 & 63);
            dst[base + fn]      = (_Float16)o0;
            dst[base + 16 + fn] = (_Float16)o1;
        }
    } else {
#pragma unroll
        for (int r = 0; r < 8; ++r) {
            int m   = r + 8 * lh;
            int row = m0 + m;
            int b   = row >> 11;
            int s   = row & (NS - 1);
            int d   = (n0 & 63) + fn;
            size_t base = (((size_t)b * NH + head) * ND + d) * NS + s;
            vT[base]           = (_Float16)c0[r];
            vT[base + 16 * NS] = (_Float16)c1[r];
        }
    }
}

// ---------------------------------------------------------------------------
// Kernel 2: causal flash attention.
//   grid = (S/16/4, B*H), block = 128; block owns 4 consecutive 16-query tiles
//   of one (b,h). K/V 32-key tiles staged to LDS with async copies, double
//   buffered; all waves follow the block-max key range to keep barriers uniform.
// ---------------------------------------------------------------------------
__global__ void __launch_bounds__(128)
attn_kernel(const _Float16* __restrict__ q,
            const _Float16* __restrict__ k,
            const _Float16* __restrict__ vT,
            _Float16* __restrict__ attn) {
    __shared__ __align__(16) _Float16 sK[2][32 * 64];  // [key-local][d]
    __shared__ __align__(16) _Float16 sV[2][64 * 32];  // [d][key-local]
    __shared__ __align__(16) _Float16 shP[4 * 16 * 32];

    int lane = threadIdx.x & 31;
    int wave = threadIdx.x >> 5;
    int qt   = blockIdx.x * 4 + wave;   // this wave's query tile (16 rows)
    int bh   = blockIdx.y;
    int fn = lane & 15, lh = lane >> 4;
    int lds0 = wave * 512;

    const _Float16* qb = q  + (size_t)bh * NS * ND;
    const _Float16* kb = k  + (size_t)bh * NS * ND;
    const _Float16* vb = vT + (size_t)bh * ND * NS;

    v16h aq0 = load_frag_g(qb, ND, qt * 16, 0);
    v16h aq1 = load_frag_g(qb, ND, qt * 16, 32);

    v8f o0 = {}, o1 = {}, o2 = {}, o3 = {};
    float mrow[8], lrow[8];
#pragma unroll
    for (int r = 0; r < 8; ++r) { mrow[r] = -1e30f; lrow[r] = 0.f; }
    const float scale = 0.125f; // 1/sqrt(64)

    int jmax_w   = (qt * 16 + 15) >> 5;                   // this wave's last key block
    int jmax_blk = ((blockIdx.x * 4 + 3) * 16 + 15) >> 5; // block max (wave 3)

    // ---- async stage of one 32-key K/V tile into LDS buffer `buf` ----------
    auto stage = [&](int j, int buf) {
        int key0 = j * 32;
        // K tile: rows key0..key0+31 are contiguous (32*64 f16 = 4KB)
#pragma unroll
        for (int c = threadIdx.x; c < 256; c += 128)
            async_copy_b128(kb + (size_t)key0 * ND + c * 8, &sK[buf][c * 8]);
        // V tile: 64 rows of 32 f16 (64B) each, row stride NS
#pragma unroll
        for (int c = threadIdx.x; c < 256; c += 128) {
            int d = c >> 2, off = (c & 3) * 8;
            async_copy_b128(vb + (size_t)d * NS + key0 + off, &sV[buf][d * 32 + off]);
        }
    };

    stage(0, 0); // prologue

    for (int j = 0; j <= jmax_blk; ++j) {
        int buf = j & 1;
        wait_async0();     // this wave's stage(j) complete
        __syncthreads();   // everyone's stage(j) complete; buf^1 free to overwrite
        if (j < jmax_blk) stage(j + 1, buf ^ 1);

        if (j <= jmax_w) {
            int key0 = j * 32;
            // S = Q @ K^T from LDS
            v16h bk00, bk01, bk10, bk11;
            LOAD_FRAG_LDS(bk00, sK[buf], ND, 0,  0);
            LOAD_FRAG_LDS(bk01, sK[buf], ND, 0,  32);
            LOAD_FRAG_LDS(bk10, sK[buf], ND, 16, 0);
            LOAD_FRAG_LDS(bk11, sK[buf], ND, 16, 32);
            v8f s0 = {}, s1 = {};
            s0 = wmma_f16(aq0, bk00, s0); s0 = wmma_f16(aq1, bk01, s0);
            s1 = wmma_f16(aq0, bk10, s1); s1 = wmma_f16(aq1, bk11, s1);

            // scale + causal mask + online softmax
#pragma unroll
            for (int r = 0; r < 8; ++r) {
                int m  = r + 8 * lh;
                int qg = qt * 16 + m;
                float x0 = (key0 + fn      <= qg) ? s0[r] * scale : -1e30f;
                float x1 = (key0 + 16 + fn <= qg) ? s1[r] * scale : -1e30f;
                float rm = fmaxf(x0, x1);
                rm = fmaxf(rm, __shfl_xor(rm, 1, 32));
                rm = fmaxf(rm, __shfl_xor(rm, 2, 32));
                rm = fmaxf(rm, __shfl_xor(rm, 4, 32));
                rm = fmaxf(rm, __shfl_xor(rm, 8, 32));
                float mn = fmaxf(mrow[r], rm);
                float al = __expf(mrow[r] - mn);
                mrow[r] = mn;
                float p0 = __expf(x0 - mn);
                float p1 = __expf(x1 - mn);
                float rs = p0 + p1;
                rs += __shfl_xor(rs, 1, 32);
                rs += __shfl_xor(rs, 2, 32);
                rs += __shfl_xor(rs, 4, 32);
                rs += __shfl_xor(rs, 8, 32);
                lrow[r] = lrow[r] * al + rs;
                o0[r] *= al; o1[r] *= al; o2[r] *= al; o3[r] *= al;
                // stage P (C-layout -> row-major 16x32 f16 in wave-private LDS)
                shP[lds0 + m * 32 + fn]      = (_Float16)p0;
                shP[lds0 + m * 32 + 16 + fn] = (_Float16)p1;
            }
            asm volatile("s_wait_dscnt 0x0" ::: "memory");

            // re-read P in A-fragment layout
            v16h ap;
            {
                int hh = lh << 3;
#pragma unroll
                for (int i = 0; i < 8; ++i) {
                    ap[i]     = shP[lds0 + fn * 32 + hh + i];
                    ap[i + 8] = shP[lds0 + fn * 32 + 16 + hh + i];
                }
            }
            // O += P @ V from LDS
            v16h bv0, bv1, bv2, bv3;
            LOAD_FRAG_LDS(bv0, sV[buf], 32, 0,  0);
            LOAD_FRAG_LDS(bv1, sV[buf], 32, 16, 0);
            LOAD_FRAG_LDS(bv2, sV[buf], 32, 32, 0);
            LOAD_FRAG_LDS(bv3, sV[buf], 32, 48, 0);
            o0 = wmma_f16(ap, bv0, o0);
            o1 = wmma_f16(ap, bv1, o1);
            o2 = wmma_f16(ap, bv2, o2);
            o3 = wmma_f16(ap, bv3, o3);
        }
    }

    // normalize and store to (B,S,H,D) f16 for the output projection
    int b = bh / NH, h = bh % NH;
#pragma unroll
    for (int r = 0; r < 8; ++r) {
        int m = r + 8 * lh;
        float inv = 1.0f / lrow[r];
        size_t rowbase = (((size_t)b * NS + qt * 16 + m) * NDM) + (size_t)h * ND;
        attn[rowbase + fn]      = (_Float16)(o0[r] * inv);
        attn[rowbase + 16 + fn] = (_Float16)(o1[r] * inv);
        attn[rowbase + 32 + fn] = (_Float16)(o2[r] * inv);
        attn[rowbase + 48 + fn] = (_Float16)(o3[r] * inv);
    }
}

// ---------------------------------------------------------------------------
// Kernel 3: output projection out = attn @ Wo^T, fp32 result.
// ---------------------------------------------------------------------------
__global__ void __launch_bounds__(128)
oproj_kernel(const _Float16* __restrict__ attnh,
             const _Float16* __restrict__ Woh,
             float* __restrict__ out) {
    int lane = threadIdx.x & 31;
    int wave = threadIdx.x >> 5;
    int m0 = blockIdx.x * 16;
    int n0 = (blockIdx.y * 4 + wave) * 32;

    v8f c0 = {}, c1 = {};
#pragma unroll 4
    for (int kk = 0; kk < NDM; kk += 32) {
        v16h a  = load_frag_g(attnh, NDM, m0, kk);
        v16h b0 = load_frag_g(Woh,   NDM, n0, kk);
        v16h b1 = load_frag_g(Woh,   NDM, n0 + 16, kk);
        c0 = wmma_f16(a, b0, c0);
        c1 = wmma_f16(a, b1, c1);
    }
    int fn = lane & 15, lh = lane >> 4;
#pragma unroll
    for (int r = 0; r < 8; ++r) {
        size_t row = (size_t)(m0 + r + 8 * lh);
        out[row * NDM + n0 + fn]      = c0[r];
        out[row * NDM + n0 + 16 + fn] = c1[r];
    }
}

// ---------------------------------------------------------------------------
// Host-side launcher
// ---------------------------------------------------------------------------
extern "C" void kernel_launch(void* const* d_in, const int* in_sizes, int n_in,
                              void* d_out, int out_size, void* d_ws, size_t ws_size,
                              hipStream_t stream) {
    const float* x   = (const float*)d_in[0];
    const int* row_ids = (const int*)d_in[1];
    const int* col_ids = (const int*)d_in[2];
    const float* Wq  = (const float*)d_in[3];
    const float* Wk  = (const float*)d_in[4];
    const float* Wv  = (const float*)d_in[5];
    const float* Wo  = (const float*)d_in[6];

    _Float16* ws = (_Float16*)d_ws;
    const size_t XN = (size_t)BS * NDM;
    const size_t WN = (size_t)NDM * NDM;
    const size_t HN = (size_t)NB * NH * NS * ND;

    _Float16* xh   = ws;
    _Float16* Wqh  = xh + XN;
    _Float16* Wkh  = Wqh + WN;
    _Float16* Wvh  = Wkh + WN;
    _Float16* Woh  = Wvh + WN;
    _Float16* qh   = Woh + WN;
    _Float16* kh   = qh + HN;
    _Float16* vTh  = kh + HN;
    _Float16* atth = vTh + HN;

    {
        int n = (int)XN;
        cvt_f16_kernel<<<(n + 255) / 256, 256, 0, stream>>>(x, xh, n);
        n = (int)WN;
        cvt_f16_kernel<<<(n + 255) / 256, 256, 0, stream>>>(Wq, Wqh, n);
        cvt_f16_kernel<<<(n + 255) / 256, 256, 0, stream>>>(Wk, Wkh, n);
        cvt_f16_kernel<<<(n + 255) / 256, 256, 0, stream>>>(Wv, Wvh, n);
        cvt_f16_kernel<<<(n + 255) / 256, 256, 0, stream>>>(Wo, Woh, n);
    }
    qkv_rope_kernel<<<dim3(BS / 16, NDM / 128, 3), 128, 0, stream>>>(
        xh, Wqh, Wkh, Wvh, qh, kh, vTh, row_ids, col_ids);
    attn_kernel<<<dim3(NS / 16 / 4, NB * NH), 128, 0, stream>>>(qh, kh, vTh, atth);
    oproj_kernel<<<dim3(BS / 16, NDM / 128), 128, 0, stream>>>(atth, Woh, (float*)d_out);
}
